// GeneralizedRingAttractorNoGain_15513421873222
// MI455X (gfx1250) — matrix operand order repaired and checked by reading, not compiled, not run on hardware
//
#include <hip/hip_runtime.h>
#include <math.h>

typedef __attribute__((ext_vector_type(16))) __bf16 v16bf;
typedef __attribute__((ext_vector_type(8)))  float  v8f;

#define NN     256
#define NBATCH 64
#define SEQ    1024
#define ADIM   2
#define M3     768          // stacked rows, interleaved per 16-row tile: Wo/Wa0/Wa1
#define CJ0    (-0.1f)
#define CJ1    (0.1f)
#define CALPHA (0.15f)

#if __has_builtin(__builtin_amdgcn_tanhf)
#define HW_TANH(x) __builtin_amdgcn_tanhf(x)
#elif __has_builtin(__builtin_amdgcn_tanh_f32)
#define HW_TANH(x) __builtin_amdgcn_tanh_f32(x)
#else
#define HW_TANH(x) tanhf(x)
#endif

union FragBf { v16bf v; uint4 q[2]; __bf16 e[16]; };

// Pure LDS barrier: wait only on DS counter (NOT storecnt), so the async bump
// global-store never gates the next recurrence step.
__device__ __forceinline__ void light_barrier() {
  asm volatile("s_wait_dscnt 0x0\n\t"
               "s_barrier_signal -1\n\t"
               "s_barrier_wait -1" ::: "memory");
}

// ---------------- kernel 0: pack weights to bf16 (tile-interleaved) ----------------
// wallI tile T = 3g+comp (T=0..47) holds rows [16g,16g+16) of {J1*Wo, Wa0, Wa1}[comp].
// => wave g's three M-tiles (3g,3g+1,3g+2) all produce the SAME output n-range.
__global__ void prep_kernel(const float* __restrict__ Wo,
                            const float* __restrict__ Wa,
                            const float* __restrict__ Wd,
                            __bf16* __restrict__ wall,
                            __bf16* __restrict__ wdb) {
  int i = blockIdx.x * blockDim.x + threadIdx.x;
  const int total = M3 * NN + NN * NN;
  if (i >= total) return;
  if (i < M3 * NN) {
    int mp = i >> 8, n = i & 255;
    int T = mp >> 4, rr = mp & 15;
    int g = T / 3, comp = T % 3;
    int row = g * 16 + rr;
    float v = (comp == 0) ? CJ1 * Wo[row * NN + n]
                          : Wa[(comp - 1) * NN * NN + row * NN + n];
    wall[i] = (__bf16)v;
  } else {
    int j = i - M3 * NN;
    wdb[j] = (__bf16)Wd[j];
  }
}

// ---------------- kernel 1: recurrent scan ----------------
// grid = 4 blocks (16 batches), block = 512 threads (16 waves).
// Wave w: A-fragments (192 VGPRs, persistent) for {J1*Wo,Wa0,Wa1} rows [16w,16w+16).
// State r lives in WMMA C/D layout (lane: b=l16, n=16w+v+8*half) -> the tanh
// update is fully in-register; only r_new -> B-layout transpose goes through a
// double-buffered 8KB LDS buffer. One light barrier per step.
__global__ __launch_bounds__(512) void scan_kernel(const float* __restrict__ action,
                                                   const float* __restrict__ r_init,
                                                   const __bf16* __restrict__ wall,
                                                   float* __restrict__ bump) {
  extern __shared__ char smem[];
  __bf16* rbf0 = (__bf16*)smem;              // [16][256] bf16, buffer 0
  __bf16* rbf1 = (__bf16*)(smem + 8192);     // [16][256] bf16, buffer 1

  const int tid  = threadIdx.x;
  const int wave = tid >> 5, lane = tid & 31;
  const int half = lane >> 4, l16 = lane & 15;
  const int bBase = blockIdx.x * 16;
  const int nBase = wave * 16 + 8 * half;    // first n of this lane's D rows

  // ---- persistent A fragments (ISA A-layout: lane = M%16, K runs of 8) ----
  FragBf A[3][8];
#pragma unroll
  for (int j = 0; j < 3; ++j) {
    const int m = (wave * 3 + j) * 16 + l16;
#pragma unroll
    for (int c = 0; c < 8; ++c) {
      const __bf16* p = wall + m * NN + c * 32 + half * 8;
      A[j][c].q[0] = *(const uint4*)p;           // K = c*32 + kb .. +7
      A[j][c].q[1] = *(const uint4*)(p + 16);    // K = c*32 + 16 + kb .. +7
    }
  }

  // ---- initial state in D layout: r[v] = r_init[b=l16][n = nBase+v] ----
  float r[8];
  {
    const float* rp = r_init + (bBase + l16) * NN + nBase;
    const float4 x0 = *(const float4*)rp;
    const float4 x1 = *(const float4*)(rp + 4);
    r[0] = x0.x; r[1] = x0.y; r[2] = x0.z; r[3] = x0.w;
    r[4] = x1.x; r[5] = x1.y; r[6] = x1.z; r[7] = x1.w;
  }
  {
    union { __bf16 e[8]; uint4 q; } pk;
#pragma unroll
    for (int i = 0; i < 8; ++i) pk.e[i] = (__bf16)r[i];
    *(uint4*)(rbf0 + l16 * NN + nBase) = pk.q;
  }
  __syncthreads();

  const float* actp = action + (size_t)(bBase + l16) * (SEQ * ADIM);
  float* bp = bump + (size_t)(bBase + l16) * SEQ * NN + nBase;

  for (int t = 0; t < SEQ; ++t) {
    const __bf16* rcur = (t & 1) ? rbf1 : rbf0;
    __bf16*       rnxt = (t & 1) ? rbf0 : rbf1;

    // early-issued per-lane action read (b = l16); consumed after the WMMAs
    const float2 av = *(const float2*)actp;
    actp += ADIM;

    v8f acc0 = {}, acc1 = {}, acc2 = {};
#pragma unroll
    for (int c = 0; c < 8; ++c) {
      FragBf Bf;  // B-layout: lane col = batch = l16, K = c*32 + half*8 + e (e<8), c*32 + 16 + half*8 + (e-8) (e>=8)
      const __bf16* p = rcur + l16 * NN + half * 8 + c * 32;
      Bf.q[0] = *(const uint4*)p;
      Bf.q[1] = *(const uint4*)(p + 16);
      acc0 = __builtin_amdgcn_wmma_f32_16x16x32_bf16(false, A[0][c].v, false, Bf.v,
                                                     (short)0, acc0, false, false);
      acc1 = __builtin_amdgcn_wmma_f32_16x16x32_bf16(false, A[1][c].v, false, Bf.v,
                                                     (short)0, acc1, false, false);
      acc2 = __builtin_amdgcn_wmma_f32_16x16x32_bf16(false, A[2][c].v, false, Bf.v,
                                                     (short)0, acc2, false, false);
    }

    // fully in-register update: acc0/1/2[v] are d0/d1/d2 for (n = nBase+v, b = l16)
    union { __bf16 e[8]; uint4 q; } pk;
#pragma unroll
    for (int v = 0; v < 8; ++v) {
      const float pre = CJ0 * r[v] + acc0[v] + av.x * acc1[v] + av.y * acc2[v];
      const float rn  = r[v] + CALPHA * (HW_TANH(pre) - r[v]);
      r[v] = rn;
      pk.e[v] = (__bf16)rn;
    }
    // transpose slice into next step's B source (double-buffered)
    *(uint4*)(rnxt + l16 * NN + nBase) = pk.q;

    // async bump store; not waited by the light barrier
    const float4 o0 = {r[0], r[1], r[2], r[3]};
    const float4 o1 = {r[4], r[5], r[6], r[7]};
    *(float4*)bp = o0;
    *(float4*)(bp + 4) = o1;
    bp += NN;

    light_barrier();
  }
}

// ---------------- kernel 2: delta7 readout + row-max normalize ----------------
// grid = 512 blocks, block = 256 threads (8 waves); each wave: 16-row strip,
// full N=256 output (16 n-tiles), K=256. W_delta7 is symmetric -> row reads.
__global__ __launch_bounds__(256) void readout_kernel(const float* __restrict__ bump,
                                                      const __bf16* __restrict__ wdb,
                                                      float* __restrict__ out) {
  extern __shared__ char smem[];
  __bf16* wlds = (__bf16*)smem;                       // [256][256] bf16 = 128 KB
  const int tid = threadIdx.x;
  {
    const uint4* src = (const uint4*)wdb;
    uint4* dst = (uint4*)wlds;
    for (int i = tid; i < (NN * NN) / 8; i += 256) dst[i] = src[i];
  }
  __syncthreads();

  const int wave = tid >> 5, lane = tid & 31;
  const int half = lane >> 4, l16 = lane & 15;
  const size_t row0 = ((size_t)blockIdx.x * 8 + wave) * 16;

  v8f acc[16];
#pragma unroll
  for (int nt = 0; nt < 16; ++nt) acc[nt] = (v8f){};

#pragma unroll
  for (int c = 0; c < 8; ++c) {
    // A fragment: bump rows f32 -> bf16, ISA A-layout
    FragBf Af;
    const float* ap = bump + (row0 + l16) * NN + c * 32 + half * 8;
    const float4 f0 = *(const float4*)(ap);
    const float4 f1 = *(const float4*)(ap + 4);
    const float4 f2 = *(const float4*)(ap + 16);
    const float4 f3 = *(const float4*)(ap + 20);
    Af.e[0]=(__bf16)f0.x; Af.e[1]=(__bf16)f0.y; Af.e[2]=(__bf16)f0.z; Af.e[3]=(__bf16)f0.w;
    Af.e[4]=(__bf16)f1.x; Af.e[5]=(__bf16)f1.y; Af.e[6]=(__bf16)f1.z; Af.e[7]=(__bf16)f1.w;
    Af.e[8]=(__bf16)f2.x; Af.e[9]=(__bf16)f2.y; Af.e[10]=(__bf16)f2.z; Af.e[11]=(__bf16)f2.w;
    Af.e[12]=(__bf16)f3.x; Af.e[13]=(__bf16)f3.y; Af.e[14]=(__bf16)f3.z; Af.e[15]=(__bf16)f3.w;
#pragma unroll
    for (int nt = 0; nt < 16; ++nt) {
      FragBf Bf;  // B element (K, n) = Wd[K][n] = Wd[n][K] (symmetric)
      const __bf16* bp = wlds + (nt * 16 + l16) * NN + c * 32 + half * 8;
      Bf.q[0] = *(const uint4*)bp;
      Bf.q[1] = *(const uint4*)(bp + 16);
      acc[nt] = __builtin_amdgcn_wmma_f32_16x16x32_bf16(false, Af.v, false, Bf.v,
                                                        (short)0, acc[nt], false, false);
    }
  }

  // row max across the 16-lane half that shares rows, then normalize + store
  float inv[8];
#pragma unroll
  for (int v = 0; v < 8; ++v) {
    float m = acc[0][v];
#pragma unroll
    for (int nt = 1; nt < 16; ++nt) m = fmaxf(m, acc[nt][v]);
    m = fmaxf(m, __shfl_xor(m, 1, 32));
    m = fmaxf(m, __shfl_xor(m, 2, 32));
    m = fmaxf(m, __shfl_xor(m, 4, 32));
    m = fmaxf(m, __shfl_xor(m, 8, 32));
    inv[v] = 1.0f / m;
  }
#pragma unroll
  for (int nt = 0; nt < 16; ++nt) {
#pragma unroll
    for (int v = 0; v < 8; ++v) {
      out[(row0 + v + 8 * half) * NN + nt * 16 + l16] = acc[nt][v] * inv[v];
    }
  }
}

// ---------------- launch ----------------
extern "C" void kernel_launch(void* const* d_in, const int* in_sizes, int n_in,
                              void* d_out, int out_size, void* d_ws, size_t ws_size,
                              hipStream_t stream) {
  (void)in_sizes; (void)n_in; (void)out_size; (void)ws_size;
  const float* action = (const float*)d_in[0];  // (64,1024,2)
  const float* r_init = (const float*)d_in[1];  // (64,256)
  const float* Wo     = (const float*)d_in[2];  // (256,256)
  const float* Wa     = (const float*)d_in[3];  // (2,256,256)
  const float* Wd     = (const float*)d_in[4];  // (256,256)

  float* out = (float*)d_out;
  const size_t OUT1 = (size_t)NBATCH * SEQ * NN;   // 16,777,216
  float* rhist = out;                // first output: r_history
  float* bump  = out + OUT1;         // second output: bump_history

  __bf16* wall = (__bf16*)d_ws;                           // 768*256 bf16 = 384 KB
  __bf16* wdb  = (__bf16*)((char*)d_ws + M3 * NN * 2);    // 256*256 bf16 = 128 KB

  prep_kernel<<<(M3 * NN + NN * NN + 255) / 256, 256, 0, stream>>>(Wo, Wa, Wd, wall, wdb);

  const size_t smem1 = 2 * 8192;             // double-buffered r (bf16, B-layout)
  scan_kernel<<<4, 512, smem1, stream>>>(action, r_init, wall, bump);

  const size_t smem2 = (size_t)NN * NN * 2;  // W_delta7 bf16 in LDS
  readout_kernel<<<NBATCH * SEQ / 128, 256, smem2, stream>>>(bump, wdb, rhist);
}
